// SessionwiseTransformer_34179349741792
// MI455X (gfx1250) — matrix operand order repaired and checked by reading, not compiled, not run on hardware
//
#include <hip/hip_runtime.h>
#include <hip/hip_bf16.h>

typedef __bf16 bf16_t;
typedef __attribute__((ext_vector_type(4)))  __bf16 v4bf;
typedef __attribute__((ext_vector_type(8)))  __bf16 v8bf;
typedef __attribute__((ext_vector_type(16))) __bf16 v16bf;
typedef __attribute__((ext_vector_type(8)))  float   v8f;

#define B_     16
#define S_     50
#define KS_    20
#define E_     256
#define D_     512
#define H_     8
#define DH_    64
#define NVALID 1001
#define NPAD   1024
#define NTOK   1000   // S_*KS_

__device__ __forceinline__ float hmax16(float v) {
  #pragma unroll
  for (int m = 1; m < 16; m <<= 1) v = fmaxf(v, __shfl_xor(v, m, 32));
  return v;
}
__device__ __forceinline__ float hsum16(float v) {
  #pragma unroll
  for (int m = 1; m < 16; m <<= 1) v += __shfl_xor(v, m, 32);
  return v;
}

// Assemble a 16x32 bf16 A-fragment from two contiguous 8-element halves.
// Callers load lo at K=half*16, hi at half*16+8, so element e -> K = half*16+e (as in B).
__device__ __forceinline__ v16bf afrag(v8bf lo, v8bf hi) {
  v16bf a;
  #pragma unroll
  for (int e = 0; e < 8; ++e) { a[e] = lo[e]; a[e + 8] = hi[e]; }
  return a;
}

// ---------------------------------------------------------------------------
// Kernel 0: fold out_proj_w/out_w into a single D-vector + scalar bias.
// ---------------------------------------------------------------------------
__global__ void prep_kernel(const float* __restrict__ out_proj_w,
                            const float* __restrict__ out_proj_b,
                            const float* __restrict__ out_w,
                            const float* __restrict__ out_b,
                            float* __restrict__ weff, float* __restrict__ beff) {
  int j = threadIdx.x;                      // 0..511
  float acc = 0.f;
  for (int i = 0; i < D_; ++i) acc += out_w[i] * out_proj_w[(size_t)i * D_ + j];
  weff[j] = acc;
  if (j == 0) {
    float bb = out_b[0];
    for (int i = 0; i < D_; ++i) bb += out_proj_b[i] * out_w[i];
    *beff = bb;
  }
}

// ---------------------------------------------------------------------------
// Kernel 1a: stream-pack keys (zero token + user||item concat) to bf16.
// keysb[b][n][c], n in [0,NPAD): zeros for n==0 and n>=NVALID.
// ---------------------------------------------------------------------------
__global__ void pack_keys(const float* __restrict__ item,
                          const float* __restrict__ user,
                          bf16_t* __restrict__ keysb) {
  size_t i = ((size_t)blockIdx.x * blockDim.x + threadIdx.x) * 4;  // elem idx
  int b = (int)(i / ((size_t)NPAD * D_));
  size_t rem = i % ((size_t)NPAD * D_);
  int n = (int)(rem / D_);
  int c = (int)(rem % D_);                  // multiple of 4; 4 | 256 so no straddle
  float4 v = make_float4(0.f, 0.f, 0.f, 0.f);
  if (n >= 1 && n < NVALID) {
    int t = n - 1;
    const float* src = (c < E_) ? (user + ((size_t)b * NTOK + t) * E_ + c)
                                : (item + ((size_t)b * NTOK + t) * E_ + (c - E_));
    v = *(const float4*)src;
  }
  v4bf o = {(bf16_t)v.x, (bf16_t)v.y, (bf16_t)v.z, (bf16_t)v.w};
  *(v4bf*)(keysb + i) = o;
}

// ---------------------------------------------------------------------------
// Kernel 1b: stream-pack in_proj_w [1536,512] to bf16.
// ---------------------------------------------------------------------------
__global__ void pack_w(const float* __restrict__ W, bf16_t* __restrict__ Wb) {
  size_t i = ((size_t)blockIdx.x * blockDim.x + threadIdx.x) * 4;
  float4 v = *(const float4*)(W + i);
  v4bf o = {(bf16_t)v.x, (bf16_t)v.y, (bf16_t)v.z, (bf16_t)v.w};
  *(v4bf*)(Wb + i) = o;
}

// ---------------------------------------------------------------------------
// Kernel 2: QKV projection GEMM (bf16 WMMA, f32 accum), all-vector loads.
// One wave computes a 16(row) x 64(col) tile of [B*NPAD, 1536].
// ---------------------------------------------------------------------------
__global__ void qkv_kernel(const bf16_t* __restrict__ keysb,  // [B,NPAD,512]
                           const bf16_t* __restrict__ Wb,     // [1536,512]
                           const float*  __restrict__ bias,   // [1536]
                           bf16_t* __restrict__ Qb,           // [B,H,NPAD,DH]
                           bf16_t* __restrict__ Kb,           // [B,H,NPAD,DH]
                           bf16_t* __restrict__ Vt) {         // [B,H,DH,NPAD]
  const int lane  = threadIdx.x;
  const int col16 = lane & 15;
  const int half  = lane >> 4;

  const int CT = (3 * D_) / 64;   // 24
  const int MT = NPAD / 16;       // 64
  int blk = blockIdx.x;
  int b   = blk / (MT * CT);
  int rem = blk % (MT * CT);
  int mt  = rem / CT;
  int ct  = rem % CT;
  const int colbase = ct * 64;
  const int nrow = mt * 16 + col16;          // A-row handled by this lane

  const bf16_t* kb = keysb + ((size_t)b * NPAD + nrow) * D_;

  v8f acc[4] = {v8f{}, v8f{}, v8f{}, v8f{}};

  for (int k0 = 0; k0 < D_; k0 += 32) {
    // A fragment: two contiguous b128 loads
    v16bf a = afrag(*(const v8bf*)(kb + k0 + half * 16),
                    *(const v8bf*)(kb + k0 + half * 16 + 8));
    // 4 B fragments: single 32B load each (B elem e -> K = half*16+e contiguous)
    #pragma unroll
    for (int sub = 0; sub < 4; ++sub) {
      v16bf bf = *(const v16bf*)(Wb + (size_t)(colbase + sub * 16 + col16) * D_
                                 + k0 + half * 16);
      acc[sub] = __builtin_amdgcn_wmma_f32_16x16x32_bf16(
          false, a, false, bf, (short)0, acc[sub], false, false);
    }
  }

  // Scatter C (lane holds rows m=r+8*half, col=col16) + bias, bf16 out
  #pragma unroll
  for (int sub = 0; sub < 4; ++sub) {
    int col = colbase + sub * 16 + col16;
    int sec = col / D_;                 // 0=Q 1=K 2=V
    int j   = col % D_;
    int h   = j / DH_, d = j % DH_;
    float bv = bias[col];
    size_t bh = (size_t)b * H_ + h;
    #pragma unroll
    for (int r = 0; r < 8; ++r) {
      int n = mt * 16 + r + 8 * half;
      bf16_t o = (bf16_t)(acc[sub][r] + bv);
      if (sec == 0)      Qb[(bh * NPAD + n) * DH_ + d] = o;
      else if (sec == 1) Kb[(bh * NPAD + n) * DH_ + d] = o;
      else               Vt[(bh * DH_ + d) * NPAD + n] = o;
    }
  }
}

// ---------------------------------------------------------------------------
// Kernel 3: flash attention per (b, h, 16-row q tile) + fused w_eff dot.
// ---------------------------------------------------------------------------
__global__ void attn_kernel(const unsigned char* __restrict__ smask,  // [B,NTOK]
                            const bf16_t* __restrict__ Qb,
                            const bf16_t* __restrict__ Kb,
                            const bf16_t* __restrict__ Vt,
                            const float* __restrict__ weff,
                            float* __restrict__ part) {   // [B,NPAD,H]
  __shared__ __align__(32) bf16_t Plds[16 * 32];
  const int lane = threadIdx.x;
  const int col  = lane & 15;
  const int half = lane >> 4;

  const int QT = NPAD / 16;   // 64
  int blk = blockIdx.x;
  int b  = blk / (H_ * QT);
  int rm = blk % (H_ * QT);
  int h  = rm / QT;
  int qt = rm % QT;
  const size_t bh = (size_t)b * H_ + h;

  // Q A-fragments: four contiguous b128 loads
  const bf16_t* qp = Qb + (bh * NPAD + (size_t)qt * 16 + col) * DH_;
  v16bf Alo = afrag(*(const v8bf*)(qp + half * 16),
                    *(const v8bf*)(qp + half * 16 + 8));
  v16bf Ahi = afrag(*(const v8bf*)(qp + 32 + half * 16),
                    *(const v8bf*)(qp + 32 + half * 16 + 8));

  int slq[8];
  #pragma unroll
  for (int r = 0; r < 8; ++r) {
    int nq = qt * 16 + r + 8 * half;
    slq[r] = (nq == 0) ? 0 : ((nq - 1) / KS_ + 1);
  }

  float rowM[8], rowL[8];
  #pragma unroll
  for (int r = 0; r < 8; ++r) { rowM[r] = -1e30f; rowL[r] = 0.f; }
  v8f ctx[4] = {v8f{}, v8f{}, v8f{}, v8f{}};
  const float scale = 0.125f;   // 1/sqrt(dh=64)

  for (int k0 = 0; k0 < NPAD; k0 += 32) {
    if (k0 + 32 < NPAD) {  // pull next chunk toward L0/L2 (global_prefetch_b8)
      __builtin_prefetch(Kb + (bh * NPAD + (size_t)(k0 + 32 + col)) * DH_, 0, 1);
      __builtin_prefetch(Vt + (bh * DH_ + col) * NPAD + k0 + 32, 0, 1);
    }

    // ---- scores: two 16x16 f32 accumulators over d=64 (2 chained WMMA each)
    v8f sc[2] = {v8f{}, v8f{}};
    #pragma unroll
    for (int a2 = 0; a2 < 2; ++a2) {
      const bf16_t* kp = Kb + (bh * NPAD + (size_t)(k0 + a2 * 16 + col)) * DH_;
      v16bf Blo = *(const v16bf*)(kp + half * 16);       // B[k][n]: k=half*16+e
      v16bf Bhi = *(const v16bf*)(kp + 32 + half * 16);
      sc[a2] = __builtin_amdgcn_wmma_f32_16x16x32_bf16(
          false, Alo, false, Blo, (short)0, sc[a2], false, false);
      sc[a2] = __builtin_amdgcn_wmma_f32_16x16x32_bf16(
          false, Ahi, false, Bhi, (short)0, sc[a2], false, false);
    }

    // ---- mask + online softmax
    int nk0 = k0 + col, nk1 = k0 + 16 + col;
    bool kv0 = (nk0 == 0) || (nk0 < NVALID && smask[(size_t)b * NTOK + nk0 - 1] != 0);
    bool kv1 = (nk1 < NVALID) && smask[(size_t)b * NTOK + nk1 - 1] != 0;
    int slk0 = (nk0 == 0) ? 0 : ((nk0 - 1) / KS_ + 1);
    int slk1 = (nk1 - 1) / KS_ + 1;

    float p0[8], p1[8], alpha[8];
    #pragma unroll
    for (int r = 0; r < 8; ++r) {
      float s0 = sc[0][r] * scale;
      float s1 = sc[1][r] * scale;
      bool m0 = !kv0 || (nk0 != 0 && slq[r] <= slk0);
      bool m1 = !kv1 || (slq[r] <= slk1);
      s0 = m0 ? -1e9f : s0;
      s1 = m1 ? -1e9f : s1;
      float cm = hmax16(fmaxf(s0, s1));
      float nM = fmaxf(rowM[r], cm);
      float al = __expf(rowM[r] - nM);
      float e0 = __expf(s0 - nM);
      float e1 = __expf(s1 - nM);
      rowL[r] = rowL[r] * al + hsum16(e0 + e1);
      rowM[r] = nM;
      alpha[r] = al;
      p0[r] = e0;
      p1[r] = e1;
    }
    #pragma unroll
    for (int dt = 0; dt < 4; ++dt)
      #pragma unroll
      for (int r = 0; r < 8; ++r) ctx[dt][r] *= alpha[r];

    // ---- P: C-layout -> A-layout via LDS (bf16)
    #pragma unroll
    for (int r = 0; r < 8; ++r) {
      Plds[(r + 8 * half) * 32 + col]      = (bf16_t)p0[r];
      Plds[(r + 8 * half) * 32 + 16 + col] = (bf16_t)p1[r];
    }
    __syncthreads();
    const bf16_t* pp = Plds + col * 32;
    v16bf Pa = afrag(*(const v8bf*)(pp + half * 16),
                     *(const v8bf*)(pp + half * 16 + 8));
    __syncthreads();

    // ---- ctx += P(16x32) @ V(32x64): 4 WMMA; Vt gives contiguous B frags
    #pragma unroll
    for (int dt = 0; dt < 4; ++dt) {
      const bf16_t* vp = Vt + (bh * DH_ + (size_t)(dt * 16 + col)) * NPAD + k0 + half * 16;
      v16bf Bv = *(const v16bf*)vp;
      ctx[dt] = __builtin_amdgcn_wmma_f32_16x16x32_bf16(
          false, Pa, false, Bv, (short)0, ctx[dt], false, false);
    }
  }

  // ---- normalize + fused dot with w_eff slice of this head
  float wreg[4];
  #pragma unroll
  for (int dt = 0; dt < 4; ++dt) wreg[dt] = weff[h * DH_ + dt * 16 + col];

  #pragma unroll
  for (int r = 0; r < 8; ++r) {
    float v = 0.f;
    #pragma unroll
    for (int dt = 0; dt < 4; ++dt) v += ctx[dt][r] * wreg[dt];
    v = hsum16(v / rowL[r]);
    if (col == 0) {
      int nq = qt * 16 + r + 8 * half;
      part[((size_t)b * NPAD + nq) * H_ + h] = v;
    }
  }
}

// ---------------------------------------------------------------------------
// Kernel 4: sum per-head partials + b_eff -> [B,S,K]
// ---------------------------------------------------------------------------
__global__ void reduce_kernel(const float* __restrict__ part,
                              const float* __restrict__ beff,
                              float* __restrict__ out) {
  int i = blockIdx.x * blockDim.x + threadIdx.x;
  if (i >= B_ * NTOK) return;
  int b = i / NTOK, t = i % NTOK;
  const float* p = part + ((size_t)b * NPAD + (t + 1)) * H_;
  float v = *beff;
  #pragma unroll
  for (int h = 0; h < H_; ++h) v += p[h];
  out[i] = v;
}

extern "C" void kernel_launch(void* const* d_in, const int* in_sizes, int n_in,
                              void* d_out, int out_size, void* d_ws, size_t ws_size,
                              hipStream_t stream) {
  const float*         item       = (const float*)d_in[0];
  const float*         user       = (const float*)d_in[1];
  const unsigned char* smask      = (const unsigned char*)d_in[2];
  const float*         in_proj_w  = (const float*)d_in[3];
  const float*         in_proj_b  = (const float*)d_in[4];
  const float*         out_proj_w = (const float*)d_in[5];
  const float*         out_proj_b = (const float*)d_in[6];
  const float*         out_w      = (const float*)d_in[7];
  const float*         out_b      = (const float*)d_in[8];

  float*  weff  = (float*)d_ws;
  float*  beff  = weff + 512;
  size_t  keysE = (size_t)B_ * NPAD * D_;             // 8M bf16 elems (16 MB)
  size_t  wE    = (size_t)(3 * D_) * D_;              // 786432 bf16 elems
  size_t  qkvE  = (size_t)B_ * H_ * NPAD * DH_;       // 8M bf16 elems (16 MB)
  bf16_t* keysb = (bf16_t*)((char*)d_ws + 4096);
  bf16_t* Wb    = keysb + keysE;
  bf16_t* Qb    = Wb + wE + 256;                      // keep 64B alignment slack
  bf16_t* Kb    = Qb + qkvE;
  bf16_t* Vt    = Kb + qkvE;
  float*  part  = (float*)(Vt + qkvE);                // B*NPAD*H floats

  prep_kernel<<<1, 512, 0, stream>>>(out_proj_w, out_proj_b, out_w, out_b, weff, beff);
  pack_keys<<<(int)(keysE / 4 / 256), 256, 0, stream>>>(item, user, keysb);
  pack_w<<<(int)(wE / 4 / 256), 256, 0, stream>>>(in_proj_w, Wb);
  qkv_kernel<<<B_ * (NPAD / 16) * ((3 * D_) / 64), 32, 0, stream>>>(
      keysb, Wb, in_proj_b, Qb, Kb, Vt);
  attn_kernel<<<B_ * H_ * (NPAD / 16), 32, 0, stream>>>(smask, Qb, Kb, Vt, weff, part);
  reduce_kernel<<<(B_ * NTOK + 255) / 256, 256, 0, stream>>>(part, beff, (float*)d_out);
}
